// MultiHeadAttention_57655640981747
// MI455X (gfx1250) — matrix operand order repaired and checked by reading, not compiled, not run on hardware
//
#include <hip/hip_runtime.h>
#include <hip/hip_bf16.h>

// Problem constants (reference: BATCH=1, SEQ=4096, D_MODEL=768, 12 heads, dh=64)
#define SEQ  4096
#define DM   768
#define NH   12
#define DH   64
#define QKVC (3 * DM)   // 2304

typedef __bf16 bf16_t;
typedef __attribute__((ext_vector_type(16))) __bf16 v16bf;
typedef __attribute__((ext_vector_type(8)))  __bf16 v8bf;
typedef __attribute__((ext_vector_type(8)))  float  v8f;

// ---------------------------------------------------------------------------
// WMMA fragment helpers (contiguous K order for A and B, wave32)
// ---------------------------------------------------------------------------

// A-matrix 16x32 bf16 fragment, source row-major with row stride lda.
// lane 0-15: M=lane, K = 0..15 (contiguous)
// lane 16-31: M=lane-16, K = 16..31
static __device__ inline v16bf load_a_frag(const bf16_t* A, int row0, int k0, int lda) {
  const int lane = threadIdx.x & 31;
  const int m    = lane & 15;
  const int kb   = (lane >> 4) << 4;                 // 0 or 16
  const bf16_t* p = A + (size_t)(row0 + m) * (size_t)lda + (size_t)(k0 + kb);
  v8bf lo = *(const v8bf*)(p);                       // K = kb .. kb+7
  v8bf hi = *(const v8bf*)(p + 8);                   // K = kb+8 .. kb+15
  return __builtin_shufflevector(lo, hi, 0,1,2,3,4,5,6,7,8,9,10,11,12,13,14,15);
}

// B-matrix 32x16 bf16 fragment. Source "Bt" stores element (k,n) at Bt[n*ldb + k]
// (K-contiguous per output column). lane 0-15: N=lane, K=0..15;
// lane 16-31: N=lane-16, K=16..31.
static __device__ inline v16bf load_b_frag(const bf16_t* Bt, int n0, int k0, int ldb) {
  const int lane = threadIdx.x & 31;
  const int n    = lane & 15;
  const int kb   = (lane >> 4) << 4;                 // 0 or 16
  const bf16_t* p = Bt + (size_t)(n0 + n) * (size_t)ldb + (size_t)(k0 + kb);
  return *(const v16bf*)p;
}

static __device__ inline v8f wmma_bf16(v16bf a, v16bf b, v8f c) {
  return __builtin_amdgcn_wmma_f32_16x16x32_bf16(false, a, false, b, (short)0, c, false, false);
}

// Row reductions inside a 16-lane half (C/D layout: a row lives in one half-wave).
static __device__ inline float half_max16(float v) {
  v = fmaxf(v, __shfl_xor(v, 1, 32));
  v = fmaxf(v, __shfl_xor(v, 2, 32));
  v = fmaxf(v, __shfl_xor(v, 4, 32));
  v = fmaxf(v, __shfl_xor(v, 8, 32));
  return v;
}
static __device__ inline float half_sum16(float v) {
  v += __shfl_xor(v, 1, 32);
  v += __shfl_xor(v, 2, 32);
  v += __shfl_xor(v, 4, 32);
  v += __shfl_xor(v, 8, 32);
  return v;
}

// ---------------------------------------------------------------------------
// Precision conversion kernels
// ---------------------------------------------------------------------------
__global__ void cvt_f32_bf16_kernel(const float* __restrict__ in,
                                    bf16_t* __restrict__ out, int n) {
  int i = blockIdx.x * blockDim.x + threadIdx.x;
  if (i < n) out[i] = (bf16_t)in[i];
}

// in[rows][cols] (f32, row-major) -> out[cols][rows] (bf16) so GEMM B-fragments
// read contiguous K.
__global__ void transpose_cvt_kernel(const float* __restrict__ in,
                                     bf16_t* __restrict__ out, int rows, int cols) {
  int c = blockIdx.x * 16 + threadIdx.x;
  int r = blockIdx.y * 16 + threadIdx.y;
  if (r < rows && c < cols)
    out[(size_t)c * rows + r] = (bf16_t)in[(size_t)r * cols + c];
}

// ---------------------------------------------------------------------------
// QKV projection: H = X @ Wqkv (4096x768 @ 768x2304).
// Register-blocked: one wave computes a 32x64 tile (2x4 accumulators = 64 VGPRs,
// spill-free): per K-step 2 A-frags + 4 B-frags feed 8 WMMAs.
// Epilogue scatters into per-head Q (scaled), K [t][dh], and V transposed [dh][t].
// ---------------------------------------------------------------------------
__global__ void __launch_bounds__(32, 1)
gemm_qkv_kernel(const bf16_t* __restrict__ Xb,
                const bf16_t* __restrict__ Wt,   // [2304][768]
                bf16_t* __restrict__ Qb,         // [NH][SEQ][DH]
                bf16_t* __restrict__ Kb,         // [NH][SEQ][DH]
                bf16_t* __restrict__ Vtb) {      // [NH][DH][SEQ]
  const int c0 = blockIdx.x * 64;      // column tile in [0,2304)
  const int t0 = blockIdx.y * 32;      // row tile in [0,4096)
  const int lane = threadIdx.x & 31;
  const int mrow = (lane >> 4) << 3;   // 0 or 8 (C-layout row base)
  const int ncol = lane & 15;

  v8f acc[2][4] = {};                  // acc[i][j]: rows t0+16i, cols c0+16j
  for (int k0 = 0; k0 < DM; k0 += 32) {
    v16bf a0 = load_a_frag(Xb, t0,      k0, DM);
    v16bf a1 = load_a_frag(Xb, t0 + 16, k0, DM);
#pragma unroll
    for (int j = 0; j < 4; ++j) {
      v16bf b = load_b_frag(Wt, c0 + 16 * j, k0, DM);
      acc[0][j] = wmma_bf16(a0, b, acc[0][j]);
      acc[1][j] = wmma_bf16(a1, b, acc[1][j]);
    }
  }

#pragma unroll
  for (int j = 0; j < 4; ++j) {
    const int c     = c0 + 16 * j + ncol;
    const int head  = c / (3 * DH);
    const int rem   = c - head * (3 * DH);
    const int which = rem >> 6;        // 0=q, 1=k, 2=v
    const int d     = rem & (DH - 1);
#pragma unroll
    for (int i = 0; i < 2; ++i) {
#pragma unroll
      for (int r = 0; r < 8; ++r) {
        const int t   = t0 + 16 * i + mrow + r;
        const float v = acc[i][j][r];
        if (which == 0)
          Qb[((size_t)head * SEQ + t) * DH + d] = (bf16_t)(v * 0.125f); // 1/sqrt(64)
        else if (which == 1)
          Kb[((size_t)head * SEQ + t) * DH + d] = (bf16_t)v;
        else
          Vtb[((size_t)head * DH + d) * SEQ + t] = (bf16_t)v;
      }
    }
  }
}

// ---------------------------------------------------------------------------
// Causal flash attention, one wave per (head, 32-query tile).
// S = Q*K^T via WMMA (dh=64 -> 2 K=32 steps); K/V B-fragments loaded once and
// reused by both 16-row query subtiles; online softmax in f32; P staged through
// LDS (C-layout -> A-layout re-tile); O += P*V via WMMA (K=32).
// ---------------------------------------------------------------------------
__global__ void __launch_bounds__(32, 1)
attn_kernel(const bf16_t* __restrict__ Q,   // [NH][SEQ][DH], pre-scaled
            const bf16_t* __restrict__ K,   // [NH][SEQ][DH]
            const bf16_t* __restrict__ Vt,  // [NH][DH][SEQ]
            bf16_t* __restrict__ AO) {      // [SEQ][DM]
  __shared__ __attribute__((aligned(32))) bf16_t pbuf[2][16 * 32];

  const int head = blockIdx.y;
  const int q0   = blockIdx.x * 32;
  const int lane = threadIdx.x & 31;
  const int mrow = (lane >> 4) << 3;
  const int ncol = lane & 15;

  const bf16_t* Qh = Q  + (size_t)head * SEQ * DH;
  const bf16_t* Kh = K  + (size_t)head * SEQ * DH;
  const bf16_t* Vh = Vt + (size_t)head * DH * SEQ;

  // Loop-invariant Q fragments: qa[qt][h] covers rows q0+16*qt, dh-halves h.
  v16bf qa[2][2];
#pragma unroll
  for (int qt = 0; qt < 2; ++qt) {
    qa[qt][0] = load_a_frag(Qh, q0 + 16 * qt, 0,  DH);
    qa[qt][1] = load_a_frag(Qh, q0 + 16 * qt, 32, DH);
  }

  v8f o[2][4] = {};                  // o[qt][j]: 16 rows x 16 d-cols
  float mstate[2][8], lstate[2][8];
#pragma unroll
  for (int qt = 0; qt < 2; ++qt)
#pragma unroll
    for (int r = 0; r < 8; ++r) { mstate[qt][r] = -1.0e30f; lstate[qt][r] = 0.0f; }

  const float NEGINF = -3.0e38f;

  for (int kb = 0; kb < q0 + 32; kb += 32) {
    // K-tile fragments, shared by both query subtiles.
    const v16bf kf0 = load_b_frag(Kh, kb,      0,  DH);
    const v16bf kf1 = load_b_frag(Kh, kb,      32, DH);
    const v16bf kf2 = load_b_frag(Kh, kb + 16, 0,  DH);
    const v16bf kf3 = load_b_frag(Kh, kb + 16, 32, DH);

    __syncthreads();   // previous iteration's LDS reads complete before rewrite
#pragma unroll
    for (int qt = 0; qt < 2; ++qt) {
      v8f s0 = {}, s1 = {};
      s0 = wmma_bf16(qa[qt][0], kf0, s0);
      s0 = wmma_bf16(qa[qt][1], kf1, s0);
      s1 = wmma_bf16(qa[qt][0], kf2, s1);
      s1 = wmma_bf16(qa[qt][1], kf3, s1);

      // Causal mask + online softmax per C-layout row.
#pragma unroll
      for (int r = 0; r < 8; ++r) {
        const int qrow = q0 + 16 * qt + mrow + r;
        float v0 = (kb + ncol      <= qrow) ? s0[r] : NEGINF;
        float v1 = (kb + 16 + ncol <= qrow) ? s1[r] : NEGINF;
        float mx = half_max16(fmaxf(v0, v1));
        const float mnew = fmaxf(mstate[qt][r], mx);
        const float e0 = __expf(v0 - mnew);
        const float e1 = __expf(v1 - mnew);
        const float rowsum = half_sum16(e0 + e1);
        const float alpha = __expf(mstate[qt][r] - mnew);
        lstate[qt][r] = lstate[qt][r] * alpha + rowsum;
        mstate[qt][r] = mnew;
#pragma unroll
        for (int j = 0; j < 4; ++j) o[qt][j][r] *= alpha;
        pbuf[qt][(mrow + r) * 32 + ncol]      = (bf16_t)e0;
        pbuf[qt][(mrow + r) * 32 + ncol + 16] = (bf16_t)e1;
      }
    }
    __syncthreads();

    // V-tile fragments (d-chunks), shared by both query subtiles.
    const v16bf vf0 = load_b_frag(Vh, 0,  kb, SEQ);
    const v16bf vf1 = load_b_frag(Vh, 16, kb, SEQ);
    const v16bf vf2 = load_b_frag(Vh, 32, kb, SEQ);
    const v16bf vf3 = load_b_frag(Vh, 48, kb, SEQ);

#pragma unroll
    for (int qt = 0; qt < 2; ++qt) {
      const v16bf pa = load_a_frag(pbuf[qt], 0, 0, 32);
      o[qt][0] = wmma_bf16(pa, vf0, o[qt][0]);
      o[qt][1] = wmma_bf16(pa, vf1, o[qt][1]);
      o[qt][2] = wmma_bf16(pa, vf2, o[qt][2]);
      o[qt][3] = wmma_bf16(pa, vf3, o[qt][3]);
    }
  }

  // Epilogue: normalize and write bf16 attention output [t][head*64+d].
#pragma unroll
  for (int qt = 0; qt < 2; ++qt) {
#pragma unroll
    for (int r = 0; r < 8; ++r) {
      const int t = q0 + 16 * qt + mrow + r;
      const float inv = 1.0f / lstate[qt][r];
      bf16_t* row = AO + (size_t)t * DM + head * DH;
      row[ncol]      = (bf16_t)(o[qt][0][r] * inv);
      row[ncol + 16] = (bf16_t)(o[qt][1][r] * inv);
      row[ncol + 32] = (bf16_t)(o[qt][2][r] * inv);
      row[ncol + 48] = (bf16_t)(o[qt][3][r] * inv);
    }
  }
}

// ---------------------------------------------------------------------------
// Output projection: out = AO @ Wproj + b, f32 output. 32x64 tile per wave.
// ---------------------------------------------------------------------------
__global__ void __launch_bounds__(32, 1)
gemm_proj_kernel(const bf16_t* __restrict__ AOb,  // [SEQ][DM]
                 const bf16_t* __restrict__ Wt,   // [DM][DM] (transposed)
                 const float*  __restrict__ bias, // [DM]
                 float* __restrict__ out) {       // [SEQ][DM]
  const int c0 = blockIdx.x * 64;
  const int t0 = blockIdx.y * 32;
  const int lane = threadIdx.x & 31;
  const int mrow = (lane >> 4) << 3;
  const int ncol = lane & 15;

  v8f acc[2][4] = {};
  for (int k0 = 0; k0 < DM; k0 += 32) {
    v16bf a0 = load_a_frag(AOb, t0,      k0, DM);
    v16bf a1 = load_a_frag(AOb, t0 + 16, k0, DM);
#pragma unroll
    for (int j = 0; j < 4; ++j) {
      v16bf b = load_b_frag(Wt, c0 + 16 * j, k0, DM);
      acc[0][j] = wmma_bf16(a0, b, acc[0][j]);
      acc[1][j] = wmma_bf16(a1, b, acc[1][j]);
    }
  }

#pragma unroll
  for (int j = 0; j < 4; ++j) {
    const int c = c0 + 16 * j + ncol;
    const float bc = bias[c];
#pragma unroll
    for (int i = 0; i < 2; ++i) {
#pragma unroll
      for (int r = 0; r < 8; ++r) {
        const int t = t0 + 16 * i + mrow + r;
        out[(size_t)t * DM + c] = acc[i][j][r] + bc;
      }
    }
  }
}

// ---------------------------------------------------------------------------
// Launch. Workspace layout (bf16 regions, ~34.5 MB total):
//   Xb[SEQ*DM] | Wqkv_t[QKVC*DM] | Wproj_t[DM*DM] | Q | K | Vt (each NH*SEQ*DH) | AO[SEQ*DM]
// ---------------------------------------------------------------------------
extern "C" void kernel_launch(void* const* d_in, const int* in_sizes, int n_in,
                              void* d_out, int out_size, void* d_ws, size_t ws_size,
                              hipStream_t stream) {
  const float* x      = (const float*)d_in[0];
  const float* w_qkv  = (const float*)d_in[1];
  const float* w_proj = (const float*)d_in[2];
  const float* b_proj = (const float*)d_in[3];
  float* out = (float*)d_out;

  bf16_t* Xb     = (bf16_t*)d_ws;
  bf16_t* Wqkvt  = Xb     + (size_t)SEQ * DM;
  bf16_t* Wprojt = Wqkvt  + (size_t)QKVC * DM;
  bf16_t* Qb     = Wprojt + (size_t)DM * DM;
  bf16_t* Kb     = Qb     + (size_t)NH * SEQ * DH;
  bf16_t* Vtb    = Kb     + (size_t)NH * SEQ * DH;
  bf16_t* AOb    = Vtb    + (size_t)NH * SEQ * DH;

  // 1) precision conversion / weight transposition
  {
    int n = SEQ * DM;
    cvt_f32_bf16_kernel<<<(n + 255) / 256, 256, 0, stream>>>(x, Xb, n);
  }
  transpose_cvt_kernel<<<dim3(QKVC / 16, DM / 16), dim3(16, 16), 0, stream>>>(
      w_qkv, Wqkvt, DM, QKVC);
  transpose_cvt_kernel<<<dim3(DM / 16, DM / 16), dim3(16, 16), 0, stream>>>(
      w_proj, Wprojt, DM, DM);

  // 2) QKV projection (32x64 register-blocked WMMA GEMM)
  gemm_qkv_kernel<<<dim3(QKVC / 64, SEQ / 32), 32, 0, stream>>>(Xb, Wqkvt, Qb, Kb, Vtb);

  // 3) causal flash attention (32 queries per wave)
  attn_kernel<<<dim3(SEQ / 32, NH), 32, 0, stream>>>(Qb, Kb, Vtb, AOb);

  // 4) output projection + bias (32x64 register-blocked WMMA GEMM)
  gemm_proj_kernel<<<dim3(DM / 64, SEQ / 32), 32, 0, stream>>>(AOb, Wprojt, b_proj, out);
}
